// NodeSelfAttention_31190052504017
// MI455X (gfx1250) — compile-verified, matrix operand order checked by reading, not run on hardware
//
#include <hip/hip_runtime.h>
#include <hip/hip_bf16.h>

// ---------------- problem constants ----------------
constexpr int B_  = 2;
constexpr int S_  = 2048;
constexpr int D_  = 512;
constexpr int H_  = 8;
constexpr int HD_ = 64;
constexpr int DFF_ = 2048;
constexpr int E_  = 65536;
constexpr int EH_ = 16;

typedef __attribute__((ext_vector_type(16))) __bf16 v16bf;
typedef __attribute__((ext_vector_type(8)))  float  v8f;
typedef __attribute__((ext_vector_type(4)))  int    int4v;

typedef __attribute__((address_space(1))) int4v global_i4;
typedef __attribute__((address_space(3))) int4v local_i4;

union F16x16 { v16bf v; uint4 q[2]; unsigned short h[16]; };
union U8x16  { uint4 u; unsigned short h[8]; };

#if __has_builtin(__builtin_amdgcn_global_load_async_to_lds_b128)
#define HAVE_ASYNC_LDS 1
#endif

static __device__ inline void async_load16(const unsigned short* g, unsigned short* l) {
#ifdef HAVE_ASYNC_LDS
    __builtin_amdgcn_global_load_async_to_lds_b128(
        (global_i4*)(uintptr_t)g,
        (local_i4*)(unsigned)(uintptr_t)l,
        0, 0);
#else
    *(uint4*)l = *(const uint4*)g;
#endif
}

static __device__ inline void wait_async_le2() {
#ifdef HAVE_ASYNC_LDS
    asm volatile("s_wait_asynccnt 2" ::: "memory");
#endif
}
static __device__ inline void wait_async_le0() {
#ifdef HAVE_ASYNC_LDS
    asm volatile("s_wait_asynccnt 0" ::: "memory");
#endif
}

static __device__ inline unsigned short f2bf_bits(float f) {
    union { float f; unsigned u; } v; v.f = f;
    unsigned u = v.u;
    unsigned r = u + 0x7FFFu + ((u >> 16) & 1u);   // round-to-nearest-even
    return (unsigned short)(r >> 16);
}

static __device__ inline float gelu_exact(float x) {
    return 0.5f * x * (1.0f + erff(x * 0.70710678118654752f));
}

static __device__ inline v8f wmma_bf16(v16bf a, v16bf b, v8f c) {
    return __builtin_amdgcn_wmma_f32_16x16x32_bf16(
        false, a, false, b, (short)0, c, false, false);
}

// ---------------- fp32 -> bf16 convert (weights) ----------------
__global__ void cvt_kernel(const float* __restrict__ in, unsigned short* __restrict__ out, int n) {
    int i = blockIdx.x * blockDim.x + threadIdx.x;
    int stride = gridDim.x * blockDim.x;
    for (; i < n; i += stride) out[i] = f2bf_bits(in[i]);
}

// ---------------- zero fill (edge bias) ----------------
__global__ void fill0_kernel(float* __restrict__ p, int n) {
    int i = blockIdx.x * blockDim.x + threadIdx.x;
    int stride = gridDim.x * blockDim.x;
    for (; i < n; i += stride) p[i] = 0.0f;
}

// ---------------- per-edge per-head MLP(1->16->1) + scatter ----------------
__global__ void edge_bias_kernel(const float* __restrict__ edge_attr,
                                 const int*  __restrict__ edge_index,
                                 const float* __restrict__ eW1, const float* __restrict__ eb1,
                                 const float* __restrict__ eW2, const float* __restrict__ eb2,
                                 float* __restrict__ bias) {
    int e = blockIdx.x * blockDim.x + threadIdx.x;
    if (e >= E_) return;
    float a = edge_attr[e];
    int r = edge_index[e];
    int c = edge_index[E_ + e];
    for (int h = 0; h < H_; ++h) {
        float s = eb2[h];
        #pragma unroll
        for (int k = 0; k < EH_; ++k) {
            float t = a * eW1[h * EH_ + k] + eb1[h * EH_ + k];
            s += gelu_exact(t) * eW2[h * EH_ + k];
        }
        bias[((size_t)h * S_ + r) * S_ + c] = s;
    }
}

// ---------------- layernorm (f32 in -> bf16 out) ----------------
__global__ __launch_bounds__(256)
void ln_kernel(const float* __restrict__ x, const float* __restrict__ g,
               const float* __restrict__ bt, unsigned short* __restrict__ out) {
    __shared__ float red[256];
    int row = blockIdx.x;
    int t = threadIdx.x;
    const float* xr = x + (size_t)row * D_;
    float a0 = xr[t], a1 = xr[t + 256];
    red[t] = a0 + a1;
    __syncthreads();
    for (int s = 128; s > 0; s >>= 1) { if (t < s) red[t] += red[t + s]; __syncthreads(); }
    float mean = red[0] * (1.0f / D_);
    __syncthreads();
    float d0 = a0 - mean, d1 = a1 - mean;
    red[t] = d0 * d0 + d1 * d1;
    __syncthreads();
    for (int s = 128; s > 0; s >>= 1) { if (t < s) red[t] += red[t + s]; __syncthreads(); }
    float rstd = rsqrtf(red[0] * (1.0f / D_) + 1e-5f);
    out[(size_t)row * D_ + t]       = f2bf_bits(d0 * rstd * g[t]       + bt[t]);
    out[(size_t)row * D_ + t + 256] = f2bf_bits(d1 * rstd * g[t + 256] + bt[t + 256]);
}

// ---------------- WMMA GEMM: C[M,N] = A[M,K](bf16) * W[K,N](bf16) + bias ----------------
// MODE 0: store bf16        MODE 1: store bf16(gelu(v))       MODE 2: store f32(v + resid)
// Double-buffered LDS, K-stage = 64, A tile filled with async-to-LDS.
template <int MODE>
__global__ __launch_bounds__(256)
void gemm_kernel(const unsigned short* __restrict__ A,
                 const unsigned short* __restrict__ W,
                 const float* __restrict__ bias,
                 const float* __restrict__ resid,
                 void* __restrict__ outp,
                 int M, int N, int K) {
    __shared__ __align__(16) unsigned short lA[2][64 * 64];   // [buf][m][k]
    __shared__ __align__(16) unsigned short lBt[2][64 * 64];  // [buf][n][k] (transposed W)
    const int m0 = blockIdx.y * 64, n0 = blockIdx.x * 64;
    const int t = threadIdx.x;
    const int wave = t >> 5, lane = t & 31, lm = lane & 15, hi = lane >> 4;
    const int sr = wave >> 1, sc0 = (wave & 1) * 2;

    // staging roles
    const int am   = t >> 2, aseg = t & 3;   // A: row am, 16B segs aseg, aseg+4
    const int wkb  = t >> 2, wseg = t & 3;   // W: row wkb, col block wseg*16

    v8f acc0 = {0,0,0,0,0,0,0,0};
    v8f acc1 = {0,0,0,0,0,0,0,0};
    uint4 breg0, breg1;

    const int nst = K >> 6;

    auto issueA = [&](int st, int buf) {
        const unsigned short* src = A + (size_t)(m0 + am) * K + st * 64;
        async_load16(src + aseg * 8,       &lA[buf][am * 64 + aseg * 8]);
        async_load16(src + (aseg + 4) * 8, &lA[buf][am * 64 + (aseg + 4) * 8]);
    };
    auto loadB = [&](int st) {
        const unsigned short* src = W + (size_t)(st * 64 + wkb) * N + n0 + wseg * 16;
        breg0 = *(const uint4*)(src);
        breg1 = *(const uint4*)(src + 8);
    };
    auto storeB = [&](int buf) {
        U8x16 u0, u1; u0.u = breg0; u1.u = breg1;
        #pragma unroll
        for (int i = 0; i < 8; ++i) lBt[buf][(wseg * 16 + i) * 64 + wkb]     = u0.h[i];
        #pragma unroll
        for (int i = 0; i < 8; ++i) lBt[buf][(wseg * 16 + 8 + i) * 64 + wkb] = u1.h[i];
    };

    issueA(0, 0);
    loadB(0);

    for (int st = 0; st < nst; ++st) {
        const int cur = st & 1, nxt = cur ^ 1;
        storeB(cur);
        if (st + 1 < nst) {
            issueA(st + 1, nxt);
            loadB(st + 1);
            wait_async_le2();
        } else {
            wait_async_le0();
        }
        __syncthreads();
        #pragma unroll
        for (int u = 0; u < 2; ++u) {
            F16x16 a, b0, b1;
            const unsigned short* pA  = &lA[cur][(sr * 16 + lm) * 64 + u * 32];
            const unsigned short* pB0 = &lBt[cur][((sc0 + 0) * 16 + lm) * 64 + u * 32];
            const unsigned short* pB1 = &lBt[cur][((sc0 + 1) * 16 + lm) * 64 + u * 32];
            a.q[0]  = *(const uint4*)(pA  + hi * 8);
            a.q[1]  = *(const uint4*)(pA  + 16 + hi * 8);
            b0.q[0] = *(const uint4*)(pB0 + hi * 8);
            b0.q[1] = *(const uint4*)(pB0 + 16 + hi * 8);
            b1.q[0] = *(const uint4*)(pB1 + hi * 8);
            b1.q[1] = *(const uint4*)(pB1 + 16 + hi * 8);
            acc0 = wmma_bf16(a.v, b0.v, acc0);
            acc1 = wmma_bf16(a.v, b1.v, acc1);
        }
        __syncthreads();
    }

    #pragma unroll
    for (int c = 0; c < 2; ++c) {
        v8f acc = c ? acc1 : acc0;
        int n = n0 + (sc0 + c) * 16 + lm;
        float bv = bias[n];
        #pragma unroll
        for (int j = 0; j < 8; ++j) {
            int m = m0 + sr * 16 + j + 8 * hi;
            float vv = acc[j] + bv;
            if (MODE == 1) vv = gelu_exact(vv);
            if (MODE == 2) {
                ((float*)outp)[(size_t)m * N + n] = vv + resid[(size_t)m * N + n];
            } else {
                ((unsigned short*)outp)[(size_t)m * N + n] = f2bf_bits(vv);
            }
        }
    }
}

// ---------------- flash attention: 1 wave per (b, h, 16 query rows) ----------------
__global__ __launch_bounds__(32)
void attn_kernel(const unsigned short* __restrict__ q,
                 const unsigned short* __restrict__ k,
                 const unsigned short* __restrict__ v,
                 const unsigned char* __restrict__ adj,
                 const float* __restrict__ bias,
                 unsigned short* __restrict__ out) {
    constexpr int NTQ = S_ / 16;
    const int id = blockIdx.x;
    const int qt = id % NTQ;
    const int h  = (id / NTQ) % H_;
    const int b  = id / (NTQ * H_);
    const int q0 = qt * 16;
    const int lane = threadIdx.x;
    const int lm = lane & 15, hi = lane >> 4;

    __shared__ __align__(16) unsigned short pLds[16 * 32];   // P tile for re-layout
    __shared__ __align__(16) unsigned short vtLds[64 * 32];  // V tile transposed [n][kb]

    // Q A-frags (16x64 split into two K=32 frags)
    F16x16 aq[2];
    {
        const unsigned short* qrow = q + ((size_t)(b * S_ + q0 + lm)) * D_ + h * HD_;
        #pragma unroll
        for (int f = 0; f < 2; ++f) {
            aq[f].q[0] = *(const uint4*)(qrow + f * 32 + hi * 8);
            aq[f].q[1] = *(const uint4*)(qrow + f * 32 + 16 + hi * 8);
        }
    }

    v8f O[4];
    const v8f z8 = {0,0,0,0,0,0,0,0};
    O[0] = z8; O[1] = z8; O[2] = z8; O[3] = z8;
    float rmax[8], rsum[8];
    #pragma unroll
    for (int j = 0; j < 8; ++j) { rmax[j] = -3.0e38f; rsum[j] = 0.0f; }

    for (int kt = 0; kt < S_ / 32; ++kt) {
        const int k0 = kt * 32;

        // prefetch next tile's K and V rows (one row per lane)
        if (kt + 1 < S_ / 32) {
            const size_t nrow = (size_t)(b * S_ + k0 + 32 + lane) * D_ + h * HD_;
            __builtin_prefetch(k + nrow, 0, 0);
            __builtin_prefetch(v + nrow, 0, 0);
        }

        v8f sc[2];
        #pragma unroll
        for (int c = 0; c < 2; ++c) {
            F16x16 bk[2];
            const unsigned short* krow = k + ((size_t)(b * S_ + k0 + c * 16 + lm)) * D_ + h * HD_;
            #pragma unroll
            for (int f = 0; f < 2; ++f) {
                bk[f].q[0] = *(const uint4*)(krow + f * 32 + hi * 8);
                bk[f].q[1] = *(const uint4*)(krow + f * 32 + 16 + hi * 8);
            }
            v8f zz = z8;
            zz = wmma_bf16(aq[0].v, bk[0].v, zz);
            zz = wmma_bf16(aq[1].v, bk[1].v, zz);
            sc[c] = zz;
        }

        // masked + biased logits
        float val[2][8];
        #pragma unroll
        for (int c = 0; c < 2; ++c)
            #pragma unroll
            for (int j = 0; j < 8; ++j) {
                int m   = j + 8 * hi;
                int key = k0 + c * 16 + lm;
                unsigned char mk = adj[((size_t)b * S_ + q0 + m) * S_ + key];
                float bv = bias[((size_t)h * S_ + q0 + m) * S_ + key];
                val[c][j] = mk ? (sc[c][j] * 0.125f + bv) : -1.0e9f;
            }

        // online softmax
        float fj[8];
        #pragma unroll
        for (int j = 0; j < 8; ++j) {
            float t = fmaxf(val[0][j], val[1][j]);
            for (int o = 1; o < 16; o <<= 1) t = fmaxf(t, __shfl_xor(t, o, 32));
            float mn = fmaxf(rmax[j], t);
            fj[j] = __expf(rmax[j] - mn);
            rmax[j] = mn;
        }
        float p[2][8];
        #pragma unroll
        for (int c = 0; c < 2; ++c)
            #pragma unroll
            for (int j = 0; j < 8; ++j) p[c][j] = __expf(val[c][j] - rmax[j]);
        #pragma unroll
        for (int j = 0; j < 8; ++j) {
            float s = p[0][j] + p[1][j];
            for (int o = 1; o < 16; o <<= 1) s += __shfl_xor(s, o, 32);
            rsum[j] = rsum[j] * fj[j] + s;
        }
        #pragma unroll
        for (int t4 = 0; t4 < 4; ++t4)
            #pragma unroll
            for (int j = 0; j < 8; ++j) O[t4][j] *= fj[j];

        // P (C-layout) -> LDS -> A-layout
        #pragma unroll
        for (int c = 0; c < 2; ++c)
            #pragma unroll
            for (int j = 0; j < 8; ++j)
                pLds[(j + 8 * hi) * 32 + c * 16 + lm] = f2bf_bits(p[c][j]);

        // stage V tile transposed: vtLds[n][kb] = V[key k0+kb][h*64+n]
        {
            const unsigned short* vrow = v + ((size_t)(b * S_ + k0 + lane)) * D_ + h * HD_;
            #pragma unroll
            for (int s8 = 0; s8 < 8; ++s8) {
                U8x16 u; u.u = *(const uint4*)(vrow + s8 * 8);
                #pragma unroll
                for (int i = 0; i < 8; ++i) vtLds[(s8 * 8 + i) * 32 + lane] = u.h[i];
            }
        }
        __syncthreads();

        F16x16 pA;
        pA.q[0] = *(const uint4*)(pLds + lm * 32 + hi * 8);
        pA.q[1] = *(const uint4*)(pLds + lm * 32 + 16 + hi * 8);
        #pragma unroll
        for (int t4 = 0; t4 < 4; ++t4) {
            F16x16 bv;
            bv.q[0] = *(const uint4*)(vtLds + (t4 * 16 + lm) * 32 + hi * 8);
            bv.q[1] = *(const uint4*)(vtLds + (t4 * 16 + lm) * 32 + 16 + hi * 8);
            O[t4] = wmma_bf16(pA.v, bv.v, O[t4]);
        }
        __syncthreads();
    }

    // normalize + store bf16 [B,S,D]
    #pragma unroll
    for (int t4 = 0; t4 < 4; ++t4)
        #pragma unroll
        for (int j = 0; j < 8; ++j) {
            int m = j + 8 * hi;
            float ov = O[t4][j] / rsum[j];
            out[((size_t)(b * S_ + q0 + m)) * D_ + h * HD_ + t4 * 16 + lm] = f2bf_bits(ov);
        }
}

// ---------------- host orchestration ----------------
extern "C" void kernel_launch(void* const* d_in, const int* in_sizes, int n_in,
                              void* d_out, int out_size, void* d_ws, size_t ws_size,
                              hipStream_t stream) {
    const float* x          = (const float*)d_in[0];
    const unsigned char* adj= (const unsigned char*)d_in[1];
    const float* edge_attr  = (const float*)d_in[2];
    const int*   edge_index = (const int*)d_in[3];
    const float* ln1_g = (const float*)d_in[4];
    const float* ln1_b = (const float*)d_in[5];
    const float* ln2_g = (const float*)d_in[6];
    const float* ln2_b = (const float*)d_in[7];
    const float* Wq = (const float*)d_in[8];   const float* bq = (const float*)d_in[9];
    const float* Wk = (const float*)d_in[10];  const float* bk = (const float*)d_in[11];
    const float* Wv = (const float*)d_in[12];  const float* bv = (const float*)d_in[13];
    const float* Wo = (const float*)d_in[14];  const float* bo = (const float*)d_in[15];
    const float* W1 = (const float*)d_in[16];  const float* b1 = (const float*)d_in[17];
    const float* W2 = (const float*)d_in[18];  const float* b2 = (const float*)d_in[19];
    const float* eW1 = (const float*)d_in[20]; const float* eb1 = (const float*)d_in[21];
    const float* eW2 = (const float*)d_in[22]; const float* eb2 = (const float*)d_in[23];

    const int M = B_ * S_;                       // 4096
    char* ws = (char*)d_ws;
    size_t off = 0;
    auto take = [&](size_t bytes) { char* p = ws + off; off += (bytes + 255) & ~(size_t)255; return p; };

    unsigned short* wqb = (unsigned short*)take((size_t)D_ * D_ * 2);
    unsigned short* wkb = (unsigned short*)take((size_t)D_ * D_ * 2);
    unsigned short* wvb = (unsigned short*)take((size_t)D_ * D_ * 2);
    unsigned short* wob = (unsigned short*)take((size_t)D_ * D_ * 2);
    unsigned short* w1b = (unsigned short*)take((size_t)D_ * DFF_ * 2);
    unsigned short* w2b = (unsigned short*)take((size_t)DFF_ * D_ * 2);
    unsigned short* nx  = (unsigned short*)take((size_t)M * D_ * 2);
    unsigned short* qb  = (unsigned short*)take((size_t)M * D_ * 2);
    unsigned short* kb  = (unsigned short*)take((size_t)M * D_ * 2);
    unsigned short* vb  = (unsigned short*)take((size_t)M * D_ * 2);
    unsigned short* ao  = (unsigned short*)take((size_t)M * D_ * 2);
    unsigned short* nx2 = (unsigned short*)take((size_t)M * D_ * 2);
    unsigned short* ffh = (unsigned short*)take((size_t)M * DFF_ * 2);
    float*          x1  = (float*)take((size_t)M * D_ * 4);
    float*          bias= (float*)take((size_t)H_ * S_ * S_ * 4);

    // weight conversion (bf16)
    cvt_kernel<<<256, 256, 0, stream>>>(Wq, wqb, D_ * D_);
    cvt_kernel<<<256, 256, 0, stream>>>(Wk, wkb, D_ * D_);
    cvt_kernel<<<256, 256, 0, stream>>>(Wv, wvb, D_ * D_);
    cvt_kernel<<<256, 256, 0, stream>>>(Wo, wob, D_ * D_);
    cvt_kernel<<<512, 256, 0, stream>>>(W1, w1b, D_ * DFF_);
    cvt_kernel<<<512, 256, 0, stream>>>(W2, w2b, DFF_ * D_);

    // dense edge bias
    fill0_kernel<<<2048, 256, 0, stream>>>(bias, H_ * S_ * S_);
    edge_bias_kernel<<<E_ / 256, 256, 0, stream>>>(edge_attr, edge_index, eW1, eb1, eW2, eb2, bias);

    // LN1 -> nx (bf16)
    ln_kernel<<<M, 256, 0, stream>>>(x, ln1_g, ln1_b, nx);

    // Q/K/V projections
    dim3 blk(256);
    dim3 gD(D_ / 64, M / 64);        // N=512
    gemm_kernel<0><<<gD, blk, 0, stream>>>(nx, wqb, bq, nullptr, qb, M, D_, D_);
    gemm_kernel<0><<<gD, blk, 0, stream>>>(nx, wkb, bk, nullptr, kb, M, D_, D_);
    gemm_kernel<0><<<gD, blk, 0, stream>>>(nx, wvb, bv, nullptr, vb, M, D_, D_);

    // attention
    attn_kernel<<<B_ * H_ * (S_ / 16), 32, 0, stream>>>(qb, kb, vb, adj, bias, ao);

    // out projection + residual (f32)
    gemm_kernel<2><<<gD, blk, 0, stream>>>(ao, wob, bo, x, x1, M, D_, D_);

    // LN2 -> nx2
    ln_kernel<<<M, 256, 0, stream>>>(x1, ln2_g, ln2_b, nx2);

    // FFN
    dim3 gF(DFF_ / 64, M / 64);      // N=2048
    gemm_kernel<1><<<gF, blk, 0, stream>>>(nx2, w1b, b1, nullptr, ffh, M, DFF_, D_);
    gemm_kernel<2><<<gD, blk, 0, stream>>>(ffh, w2b, b2, x1, (float*)d_out, M, D_, DFF_);
}
